// TransductivePrototypicalNetwork_15006615733039
// MI455X (gfx1250) — compile-verified, matrix operand order checked by reading, not run on hardware
//
#include <hip/hip_runtime.h>
#include <math.h>

// ---------------------------------------------------------------------------
// Transductive Prototypical Network for MI455X (gfx1250, wave32, WMMA).
// Heavy GEMMs run as bf16 v_wmma_f32_16x16x32_bf16 with fp32 accumulation.
// Inner loops hoist all 6 fragments (2xA, 4xB) before an 8-WMMA burst so the
// ds_loads clause together under one s_wait_dscnt instead of serializing.
// ---------------------------------------------------------------------------

typedef __attribute__((ext_vector_type(16))) __bf16 v16bf;
typedef __attribute__((ext_vector_type(8)))  __bf16 v8bf;
typedef __attribute__((ext_vector_type(4)))  __bf16 v4bf;
typedef __attribute__((ext_vector_type(8)))  float  v8f;

#define N_CLASSES 64
#define D_IN      2048
#define D_EMB     1024
#define NS        1280
#define NQ        65536
#define NALL      (NS + NQ)
#define T_STEPS   5

// ---- WMMA fragment load helpers (ISA 7.12.2 16-bit A/B layouts) -----------
// A (16x32, bf16): lane L holds row M=L&15; K-chunks [kk..kk+7] and
// [kk+16..kk+23] with kk = (L&16)?8:0. B (32x16) is the symmetric layout with
// column N=L&15.  Tiles in LDS are stored row/col-major with K stride 32.
static __device__ __forceinline__ v16bf load_frag16(const __bf16* base, int lane) {
  const int half = (lane >> 4) & 1;
  const int idx  = lane & 15;
  const __bf16* p = base + idx * 32 + half * 8;
  v8bf lo = *(const v8bf*)(p);
  v8bf hi = *(const v8bf*)(p + 16);
  return __builtin_shufflevector(lo, hi, 0,1,2,3,4,5,6,7,8,9,10,11,12,13,14,15);
}

static __device__ __forceinline__ v8bf cvt8(const float* f) {
  v8bf r;
#pragma unroll
  for (int i = 0; i < 8; ++i) r[i] = (__bf16)f[i];
  return r;
}

static __device__ __forceinline__ v8f wmma_bf16(v16bf a, v16bf b, v8f c) {
  return __builtin_amdgcn_wmma_f32_16x16x32_bf16(false, a, false, b,
                                                 (short)0, c, false, false);
}

// ---- W_enc fp32 [2048][1024] -> bf16 transposed Wt [1024][2048] ------------
__global__ __launch_bounds__(256) void convert_w_kernel(
    const float* __restrict__ W, __bf16* __restrict__ Wt) {
  const int idx = blockIdx.x * 256 + threadIdx.x;   // 2048*1024 threads
  const int k = idx >> 10;
  const int n = idx & 1023;
  Wt[(size_t)n * D_IN + k] = (__bf16)W[idx];
}

__global__ void copy_labels_kernel(const int* __restrict__ s, int* __restrict__ l) {
  const int i = blockIdx.x * 256 + threadIdx.x;
  if (i < NS) l[i] = s[i];
}

__global__ void zero_kernel(float* __restrict__ p, int n) {
  const int i = blockIdx.x * 256 + threadIdx.x;
  if (i < n) p[i] = 0.0f;
}

// ---- Encoder GEMM: emb[m][n] = A[m][:] @ W[:,n] + b[n], stored bf16 --------
// 128x128 tile per block, 8 waves, each wave 32x64 = 2x4 WMMA frags, K=2048.
__global__ __launch_bounds__(256) void encode_gemm_kernel(
    const float* __restrict__ A, const __bf16* __restrict__ Wt,
    const float* __restrict__ bias, __bf16* __restrict__ emb, int rowOffset) {
  __shared__ __align__(16) __bf16 As[128 * 32];   // [m][k] 8 KB
  __shared__ __align__(16) __bf16 Bs[128 * 32];   // [n][k] 8 KB
  const int tid  = threadIdx.x;
  const int lane = tid & 31;
  const int wid  = tid >> 5;
  const int half = (lane >> 4) & 1;
  const int lx   = lane & 15;
  const int m0 = blockIdx.x * 128;
  const int n0 = blockIdx.y * 128;
  const int wm = (wid & 3) * 32;
  const int wn = (wid >> 2) * 64;

  const int lrow = tid >> 1;          // 0..127
  const int lk   = (tid & 1) * 16;    // 0 or 16

  v8f c[2][4] = {};

  for (int k0 = 0; k0 < D_IN; k0 += 32) {
    // A tile: fp32 -> bf16 into LDS
    const float* ap = A + (size_t)(m0 + lrow) * D_IN + k0 + lk;
    float fa[16];
#pragma unroll
    for (int i = 0; i < 16; i += 4) {
      float4 v = *(const float4*)(ap + i);
      fa[i] = v.x; fa[i + 1] = v.y; fa[i + 2] = v.z; fa[i + 3] = v.w;
    }
    v8bf* adst = (v8bf*)&As[lrow * 32 + lk];
    adst[0] = cvt8(fa);
    adst[1] = cvt8(fa + 8);

    // B tile from pre-transposed Wt (already K-contiguous)
    const v8bf* bp = (const v8bf*)(Wt + (size_t)(n0 + lrow) * D_IN + k0 + lk);
    v8bf b0v = bp[0], b1v = bp[1];
    v8bf* bdst = (v8bf*)&Bs[lrow * 32 + lk];
    bdst[0] = b0v; bdst[1] = b1v;

    if (k0 + 32 < D_IN)
      __builtin_prefetch(ap + 32, 0, 1);

    __syncthreads();

    // Hoist ALL fragments so ds_loads clause under one wait, then burst WMMAs.
    v16bf a0 = load_frag16(&As[(wm + 0)  * 32], lane);
    v16bf a1 = load_frag16(&As[(wm + 16) * 32], lane);
    v16bf b0 = load_frag16(&Bs[(wn + 0)  * 32], lane);
    v16bf b1 = load_frag16(&Bs[(wn + 16) * 32], lane);
    v16bf b2 = load_frag16(&Bs[(wn + 32) * 32], lane);
    v16bf b3 = load_frag16(&Bs[(wn + 48) * 32], lane);

    c[0][0] = wmma_bf16(a0, b0, c[0][0]);
    c[0][1] = wmma_bf16(a0, b1, c[0][1]);
    c[0][2] = wmma_bf16(a0, b2, c[0][2]);
    c[0][3] = wmma_bf16(a0, b3, c[0][3]);
    c[1][0] = wmma_bf16(a1, b0, c[1][0]);
    c[1][1] = wmma_bf16(a1, b1, c[1][1]);
    c[1][2] = wmma_bf16(a1, b2, c[1][2]);
    c[1][3] = wmma_bf16(a1, b3, c[1][3]);

    __syncthreads();
  }

  // Epilogue: add bias, store bf16.  C layout: n = lane&15, m = r + 8*(lane>>4).
#pragma unroll
  for (int in = 0; in < 4; ++in) {
    const int n = n0 + wn + in * 16 + lx;
    const float bn = bias[n];
#pragma unroll
    for (int im = 0; im < 2; ++im) {
#pragma unroll
      for (int r = 0; r < 8; ++r) {
        const int m = m0 + wm + im * 16 + r + 8 * half;
        emb[(size_t)(rowOffset + m) * D_EMB + n] = (__bf16)(c[im][in][r] + bn);
      }
    }
  }
}

// ---- Row squared-norms of the stored bf16 embeddings -----------------------
__global__ __launch_bounds__(256) void row_norm_kernel(
    const __bf16* __restrict__ emb, float* __restrict__ xnorm) {
  const int wid = threadIdx.x >> 5, lane = threadIdx.x & 31;
  const int row = blockIdx.x * 8 + wid;
  const __bf16* e = emb + (size_t)row * D_EMB;
  float s = 0.0f;
#pragma unroll
  for (int t = 0; t < 4; ++t) {
    v8bf v = *(const v8bf*)(e + t * 256 + lane * 8);
#pragma unroll
    for (int i = 0; i < 8; ++i) { float f = (float)v[i]; s = fmaf(f, f, s); }
  }
#pragma unroll
  for (int off = 16; off; off >>= 1) s += __shfl_xor(s, off, 32);
  if (lane == 0) xnorm[row] = s;
}

// ---- Prototype accumulation: all 64 classes live in one WGP's LDS ----------
// 64*1024 fp32 accumulators = 256 KB dynamic LDS (WGP has 320 KB).
// Each thread owns 4 fixed dims -> plain LDS += (no atomics in the hot loop).
__global__ __launch_bounds__(256) void proto_accum_kernel(
    const __bf16* __restrict__ emb, const int* __restrict__ labels,
    float* __restrict__ sums, float* __restrict__ counts, int nrows) {
  extern __shared__ float lacc[];                  // [64*1024] + [64]
  float* lcnt = lacc + N_CLASSES * D_EMB;
  for (int i = threadIdx.x; i < N_CLASSES * D_EMB + N_CLASSES; i += 256)
    lacc[i] = 0.0f;
  __syncthreads();

  const int d0 = threadIdx.x * 4;                  // this thread's 4 dims
  for (int row = blockIdx.x; row < nrows; row += gridDim.x) {
    const int lab = labels[row];
    v4bf v = *(const v4bf*)(emb + (size_t)row * D_EMB + d0);
    float* a = lacc + lab * D_EMB + d0;
    a[0] += (float)v[0]; a[1] += (float)v[1];
    a[2] += (float)v[2]; a[3] += (float)v[3];
    if (threadIdx.x == 0) lcnt[lab] += 1.0f;
  }
  __syncthreads();

  for (int i = threadIdx.x; i < N_CLASSES * D_EMB; i += 256) {
    const float v = lacc[i];
    if (v != 0.0f) atomicAdd(&sums[i], v);
  }
  if (threadIdx.x < N_CLASSES) {
    const float v = lcnt[threadIdx.x];
    if (v != 0.0f) atomicAdd(&counts[threadIdx.x], v);
  }
}

// ---- Prototype finalize: mean, bf16 copy (K-contiguous), squared norm ------
__global__ __launch_bounds__(256) void proto_finalize_kernel(
    const float* __restrict__ sums, const float* __restrict__ counts,
    __bf16* __restrict__ protoT, float* __restrict__ pnorm) {
  __shared__ float red[256];
  const int cls = blockIdx.x;
  const float inv = 1.0f / fmaxf(counts[cls], 1.0f);
  float s = 0.0f;
  for (int d = threadIdx.x; d < D_EMB; d += 256) {
    const float m = sums[cls * D_EMB + d] * inv;
    const __bf16 mb = (__bf16)m;
    protoT[cls * D_EMB + d] = mb;
    const float mf = (float)mb;
    s = fmaf(mf, mf, s);
  }
  red[threadIdx.x] = s;
  __syncthreads();
  for (int off = 128; off; off >>= 1) {
    if (threadIdx.x < off) red[threadIdx.x] += red[threadIdx.x + off];
    __syncthreads();
  }
  if (threadIdx.x == 0) pnorm[cls] = red[0];
}

// ---- Distance GEMM + fused argmin (mode 0) / logits (mode 1) ---------------
// 256x64 tile per block (8 waves x 32 rows), K = 1024.
__global__ __launch_bounds__(256) void dist_kernel(
    const __bf16* __restrict__ emb, const __bf16* __restrict__ protoT,
    const float* __restrict__ xnorm, const float* __restrict__ pnorm,
    int mode, int* __restrict__ labels, float* __restrict__ out) {
  __shared__ __align__(16) __bf16 As[256 * 32];   // 16 KB
  __shared__ __align__(16) __bf16 Bs[64 * 32];    //  4 KB
  const int tid  = threadIdx.x;
  const int lane = tid & 31;
  const int wid  = tid >> 5;
  const int half = (lane >> 4) & 1;
  const int lx   = lane & 15;
  const int m0 = blockIdx.x * 256;
  const int wm = wid * 32;
  const __bf16* qemb = emb + (size_t)NS * D_EMB;

  v8f c[2][4] = {};

  for (int k0 = 0; k0 < D_EMB; k0 += 32) {
    const v8bf* ap = (const v8bf*)(qemb + (size_t)(m0 + tid) * D_EMB + k0);
    v8bf a0v = ap[0], a1v = ap[1], a2v = ap[2], a3v = ap[3];
    v8bf* adst = (v8bf*)&As[tid * 32];
    adst[0] = a0v; adst[1] = a1v; adst[2] = a2v; adst[3] = a3v;
    if (tid < 128) {
      const int br = tid >> 1, bk = (tid & 1) * 16;
      const v8bf* bp = (const v8bf*)(protoT + (size_t)br * D_EMB + k0 + bk);
      v8bf b0v = bp[0], b1v = bp[1];
      v8bf* bdst = (v8bf*)&Bs[br * 32 + bk];
      bdst[0] = b0v; bdst[1] = b1v;
    }
    if (k0 + 32 < D_EMB)
      __builtin_prefetch(qemb + (size_t)(m0 + tid) * D_EMB + k0 + 32, 0, 1);
    __syncthreads();

    v16bf a0 = load_frag16(&As[(wm + 0)  * 32], lane);
    v16bf a1 = load_frag16(&As[(wm + 16) * 32], lane);
    v16bf b0 = load_frag16(&Bs[0  * 32], lane);
    v16bf b1 = load_frag16(&Bs[16 * 32], lane);
    v16bf b2 = load_frag16(&Bs[32 * 32], lane);
    v16bf b3 = load_frag16(&Bs[48 * 32], lane);

    c[0][0] = wmma_bf16(a0, b0, c[0][0]);
    c[0][1] = wmma_bf16(a0, b1, c[0][1]);
    c[0][2] = wmma_bf16(a0, b2, c[0][2]);
    c[0][3] = wmma_bf16(a0, b3, c[0][3]);
    c[1][0] = wmma_bf16(a1, b0, c[1][0]);
    c[1][1] = wmma_bf16(a1, b1, c[1][1]);
    c[1][2] = wmma_bf16(a1, b2, c[1][2]);
    c[1][3] = wmma_bf16(a1, b3, c[1][3]);

    __syncthreads();
  }

  float pn[4];
#pragma unroll
  for (int in = 0; in < 4; ++in) pn[in] = pnorm[in * 16 + lx];

#pragma unroll
  for (int im = 0; im < 2; ++im) {
#pragma unroll
    for (int r = 0; r < 8; ++r) {
      const int m = m0 + wm + im * 16 + r + 8 * half;   // query row
      const float xn = xnorm[NS + m];
      if (mode == 0) {
        float best = 3.4e38f; int bi = 0;
#pragma unroll
        for (int in = 0; in < 4; ++in) {
          const float d2 = xn + pn[in] - 2.0f * c[im][in][r];
          const int n = in * 16 + lx;
          if (d2 < best) { best = d2; bi = n; }
        }
        // reduce across the 16 lanes of this half (xor 1..8 stays in-half)
#pragma unroll
        for (int off = 1; off < 16; off <<= 1) {
          const float ob = __shfl_xor(best, off, 32);
          const int   oi = __shfl_xor(bi, off, 32);
          if (ob < best || (ob == best && oi < bi)) { best = ob; bi = oi; }
        }
        if (lx == 0) labels[NS + m] = bi;
      } else {
#pragma unroll
        for (int in = 0; in < 4; ++in) {
          const int n = in * 16 + lx;
          const float d2 = xn + pn[in] - 2.0f * c[im][in][r];
          out[(size_t)m * N_CLASSES + n] = -sqrtf(fmaxf(d2, 1e-12f));
        }
      }
    }
  }
}

// ---------------------------------------------------------------------------
extern "C" void kernel_launch(void* const* d_in, const int* in_sizes, int n_in,
                              void* d_out, int out_size, void* d_ws, size_t ws_size,
                              hipStream_t stream) {
  (void)in_sizes; (void)n_in; (void)out_size; (void)ws_size;
  const float* support = (const float*)d_in[0];   // [1280, 2048]
  const float* query   = (const float*)d_in[1];   // [65536, 2048]
  const int*   slab    = (const int*)d_in[2];     // [1280]
  const float* W       = (const float*)d_in[3];   // [2048, 1024]
  const float* bias    = (const float*)d_in[4];   // [1024]
  float* out = (float*)d_out;                     // [65536, 64]

  char* base = (char*)d_ws;
  size_t off = 0;
  auto take = [&](size_t bytes) -> char* {
    char* p = base + off;
    off = (off + bytes + 255) & ~(size_t)255;
    return p;
  };
  __bf16* Wt     = (__bf16*)take((size_t)D_IN * D_EMB * 2);        //   4 MB
  __bf16* emb    = (__bf16*)take((size_t)NALL * D_EMB * 2);        // 137 MB
  float*  xnorm  = (float*)take((size_t)NALL * 4);
  int*    labels = (int*)take((size_t)NALL * 4);
  float*  sums   = (float*)take((size_t)(N_CLASSES * D_EMB + N_CLASSES) * 4);
  float*  counts = sums + N_CLASSES * D_EMB;
  __bf16* protoT = (__bf16*)take((size_t)N_CLASSES * D_EMB * 2);
  float*  pnorm  = (float*)take((size_t)N_CLASSES * 4);

  const int zn = N_CLASSES * D_EMB + N_CLASSES;
  const size_t accum_lds = (size_t)zn * sizeof(float);   // 256 KB + 256 B

  // One-time prep
  convert_w_kernel<<<(D_IN * D_EMB) / 256, 256, 0, stream>>>(W, Wt);
  copy_labels_kernel<<<(NS + 255) / 256, 256, 0, stream>>>(slab, labels);

  // Encode support + query (bf16 WMMA, fp32 accumulate, +bias)
  encode_gemm_kernel<<<dim3(NS / 128, D_EMB / 128), 256, 0, stream>>>(
      support, Wt, bias, emb, 0);
  encode_gemm_kernel<<<dim3(NQ / 128, D_EMB / 128), 256, 0, stream>>>(
      query, Wt, bias, emb, NS);
  row_norm_kernel<<<NALL / 8, 256, 0, stream>>>(emb, xnorm);

  // Initial prototypes from support only
  zero_kernel<<<(zn + 255) / 256, 256, 0, stream>>>(sums, zn);
  proto_accum_kernel<<<256, 256, accum_lds, stream>>>(emb, labels, sums, counts, NS);
  proto_finalize_kernel<<<N_CLASSES, 256, 0, stream>>>(sums, counts, protoT, pnorm);

  // Transductive refinement
  for (int step = 0; step < T_STEPS; ++step) {
    dist_kernel<<<NQ / 256, 256, 0, stream>>>(emb, protoT, xnorm, pnorm,
                                              0, labels, nullptr);
    zero_kernel<<<(zn + 255) / 256, 256, 0, stream>>>(sums, zn);
    proto_accum_kernel<<<256, 256, accum_lds, stream>>>(emb, labels, sums, counts, NALL);
    proto_finalize_kernel<<<N_CLASSES, 256, 0, stream>>>(sums, counts, protoT, pnorm);
  }

  // Final logits
  dist_kernel<<<NQ / 256, 256, 0, stream>>>(emb, protoT, xnorm, pnorm,
                                            1, nullptr, out);
}
